// Cross_MultiViewAtt_26637387170183
// MI455X (gfx1250) — matrix operand order repaired and checked by reading, not compiled, not run on hardware
//
#include <hip/hip_runtime.h>

// ---------------------------------------------------------------------------
// CDNA5 (gfx1250, wave32) implementation of the Cross_MultiViewAtt module.
// All matmuls run through v_wmma_f32_16x16x32_bf16 (bf16 in, f32 accum).
// ---------------------------------------------------------------------------

typedef __bf16 bf16;
typedef __attribute__((ext_vector_type(16))) __bf16 v16bf;
typedef __attribute__((ext_vector_type(8)))  __bf16 bf16x8;
typedef __attribute__((ext_vector_type(8)))  float   v8f;

union FragU { v16bf v; bf16x8 h[2]; };

// ===========================================================================
// Batched GEMM:  Y[b, m, n] = act( sum_k A[b,m,k] * Wt[b,n,k] + bias )
// A: bf16 row-major (lda = K-stride), Wt: bf16 row-major (N rows x K),
// Y: f32 (ldy). Workgroup = 256 thr = 8 waves; tile 128x64, BK = 64 (2 WMMA
// K-steps per stage). Each wave: 32x32 output = 2x2 WMMA 16x16x32 frags.
// GUARDED=false fast path: requires M%128==0, N%64==0, K%64==0.
// GUARDED=true: requires only K%8==0 (true for every call here); row OOB is
// handled by clamping the row index (stores are guarded), K OOB by a zero
// vector -- no per-element scalar fallback.
// ===========================================================================
template <bool GUARDED>
__global__ __launch_bounds__(256) void k_gemm(
    const bf16* __restrict__ A,  long sAb, int lda,
    const bf16* __restrict__ Wt, long sWb, int ldw,
    float* __restrict__ Y,       long sYb, int ldy,
    const float* __restrict__ bias, int bias_mode, int do_relu,
    int M, int N, int K)
{
    __shared__ __align__(16) bf16 As[128 * 64];
    __shared__ __align__(16) bf16 Bs[64 * 64];

    const int bz = blockIdx.z;
    const bf16* Ab = A  + (long)bz * sAb;
    const bf16* Wb = Wt + (long)bz * sWb;
    float*      Yb = Y  + (long)bz * sYb;

    const int m0 = blockIdx.y * 128;
    const int n0 = blockIdx.x * 64;
    const int t    = threadIdx.x;
    const int lane = t & 31;
    const int wid  = t >> 5;
    const int wm   = wid & 3;        // 4 wave-rows  -> 128 M
    const int wn   = wid >> 2;       // 2 wave-cols  -> 64  N
    const int r16  = lane & 15;
    const int hi   = lane >> 4;      // 0 or 1

    v8f acc[2][2];
    #pragma unroll
    for (int i = 0; i < 2; ++i)
        #pragma unroll
        for (int j = 0; j < 2; ++j)
            #pragma unroll
            for (int e = 0; e < 8; ++e) acc[i][j][e] = 0.f;

    for (int k0 = 0; k0 < K; k0 += 64) {
        // ---- stage A tile (128x64): 1024 chunks of 8 bf16, 4 per thread ----
        #pragma unroll
        for (int i = 0; i < 4; ++i) {
            int c = t + (i << 8);
            int row = c >> 3, seg = (c & 7) << 3;
            bf16x8 v;
            if (!GUARDED) {
                const bf16* p = Ab + (long)(m0 + row) * lda + (k0 + seg);
                v = *(const bf16x8*)p;
                if (i == 0) __builtin_prefetch(p + 64, 0, 1);   // next K-slice
            } else {
                int gr = m0 + row;  if (gr >= M) gr = M - 1;    // clamp; stores guarded
                int gk = k0 + seg;
                if (gk + 8 <= K) {                               // K % 8 == 0 always
                    v = *(const bf16x8*)(Ab + (long)gr * lda + gk);
                } else {
                    #pragma unroll
                    for (int j = 0; j < 8; ++j) ((bf16*)&v)[j] = (bf16)0.f;
                }
            }
            *(bf16x8*)&As[(row << 6) + seg] = v;
        }
        // ---- stage B tile (64x64): 512 chunks, 2 per thread ----
        #pragma unroll
        for (int i = 0; i < 2; ++i) {
            int c = t + (i << 8);
            int row = c >> 3, seg = (c & 7) << 3;
            bf16x8 v;
            if (!GUARDED) {
                const bf16* p = Wb + (long)(n0 + row) * ldw + (k0 + seg);
                v = *(const bf16x8*)p;
                if (i == 0) __builtin_prefetch(p + 64, 0, 1);
            } else {
                int gr = n0 + row;  if (gr >= N) gr = N - 1;
                int gk = k0 + seg;
                if (gk + 8 <= K) {
                    v = *(const bf16x8*)(Wb + (long)gr * ldw + gk);
                } else {
                    #pragma unroll
                    for (int j = 0; j < 8; ++j) ((bf16*)&v)[j] = (bf16)0.f;
                }
            }
            *(bf16x8*)&Bs[(row << 6) + seg] = v;
        }
        __syncthreads();

        // ---- two WMMA K-steps from LDS (ISA 7.12.2 fragment layouts) ----
        #pragma unroll
        for (int s = 0; s < 2; ++s) {
            const int kb = s << 5;
            FragU af[2], bfg[2];
            const int koffA = kb + (hi << 3);  // A: lane<16 -> K0..7 & K16..23
            #pragma unroll
            for (int mi = 0; mi < 2; ++mi) {
                int row = (wm << 5) + (mi << 4) + r16;
                af[mi].h[0] = *(const bf16x8*)&As[(row << 6) + koffA];
                af[mi].h[1] = *(const bf16x8*)&As[(row << 6) + koffA + 16];
            }
            const int koffB = kb + (hi << 3);  // B: lane<16 -> K0..7 & K16..23
            #pragma unroll
            for (int ni = 0; ni < 2; ++ni) {
                int col = (wn << 5) + (ni << 4) + r16;
                bfg[ni].h[0] = *(const bf16x8*)&Bs[(col << 6) + koffB];
                bfg[ni].h[1] = *(const bf16x8*)&Bs[(col << 6) + koffB + 16];
            }
            #pragma unroll
            for (int mi = 0; mi < 2; ++mi)
                #pragma unroll
                for (int ni = 0; ni < 2; ++ni)
                    acc[mi][ni] = __builtin_amdgcn_wmma_f32_16x16x32_bf16(
                        false, af[mi].v, false, bfg[ni].v,
                        (short)0, acc[mi][ni], false, false);
        }
        __syncthreads();
    }

    // ---- epilogue: C/D layout: VGPR r -> M = r + 8*(lane>=16), N = lane&15 ----
    const int rofs = hi << 3;
    #pragma unroll
    for (int mi = 0; mi < 2; ++mi) {
        #pragma unroll
        for (int ni = 0; ni < 2; ++ni) {
            int col = n0 + (wn << 5) + (ni << 4) + r16;
            if (GUARDED && col >= N) continue;
            #pragma unroll
            for (int r = 0; r < 8; ++r) {
                int row = m0 + (wm << 5) + (mi << 4) + rofs + r;
                if (GUARDED && row >= M) continue;
                float v = acc[mi][ni][r];
                if (bias_mode == 1)      v += bias[col];
                else if (bias_mode == 2) v += bias[row];
                if (do_relu) v = v > 0.f ? v : 0.f;
                Yb[(long)row * ldy + col] = v;
            }
        }
    }
}

// ===========================================================================
// Fused residual-add + LayerNorm (C = 512). Writes f32 + bf16 copies.
// One 256-thread block per token row.
// ===========================================================================
__global__ __launch_bounds__(256) void k_add_ln(
    const float* __restrict__ a, const float* __restrict__ b,
    const float* __restrict__ g, const float* __restrict__ be,
    float* __restrict__ outf, bf16* __restrict__ outb, int C)
{
    __shared__ float xs[512];
    __shared__ float red[256];
    long row = blockIdx.x;
    const float* ar = a + row * C;
    const float* br = b + row * C;
    int t = threadIdx.x;
    float s = 0.f;
    for (int i = t; i < C; i += 256) { float v = ar[i] + br[i]; xs[i] = v; s += v; }
    red[t] = s; __syncthreads();
    for (int o = 128; o > 0; o >>= 1) { if (t < o) red[t] += red[t + o]; __syncthreads(); }
    float mean = red[0] / C; __syncthreads();
    float s2 = 0.f;
    for (int i = t; i < C; i += 256) { float d = xs[i] - mean; s2 += d * d; }
    red[t] = s2; __syncthreads();
    for (int o = 128; o > 0; o >>= 1) { if (t < o) red[t] += red[t + o]; __syncthreads(); }
    float rstd = rsqrtf(red[0] / C + 1e-5f);
    for (int i = t; i < C; i += 256) {
        float v = (xs[i] - mean) * rstd * g[i] + be[i];
        outf[row * C + i] = v;
        outb[row * C + i] = (bf16)v;
    }
}

// ===========================================================================
// Row softmax with scale; emits bf16 probabilities. len <= 1024.
// ===========================================================================
__global__ __launch_bounds__(256) void k_softmax(
    const float* __restrict__ S, bf16* __restrict__ P, int len, float scale)
{
    __shared__ float xs[1024];
    __shared__ float red[256];
    long row = blockIdx.x;
    const float* sr = S + row * (long)len;
    int t = threadIdx.x;
    float mx = -3.4e38f;
    for (int i = t; i < len; i += 256) { float v = sr[i] * scale; xs[i] = v; mx = fmaxf(mx, v); }
    red[t] = mx; __syncthreads();
    for (int o = 128; o > 0; o >>= 1) { if (t < o) red[t] = fmaxf(red[t], red[t + o]); __syncthreads(); }
    mx = red[0]; __syncthreads();
    float s = 0.f;
    for (int i = t; i < len; i += 256) { float e = __expf(xs[i] - mx); xs[i] = e; s += e; }
    red[t] = s; __syncthreads();
    for (int o = 128; o > 0; o >>= 1) { if (t < o) red[t] += red[t + o]; __syncthreads(); }
    float inv = 1.f / red[0];
    for (int i = t; i < len; i += 256) P[row * (long)len + i] = (bf16)(xs[i] * inv);
}

// ===========================================================================
// Elementwise / layout kernels (grid-stride)
// ===========================================================================
__global__ void k_f32_to_bf16(const float* __restrict__ s, bf16* __restrict__ d, long n) {
    for (long i = blockIdx.x * 256L + threadIdx.x; i < n; i += gridDim.x * 256L)
        d[i] = (bf16)s[i];
}

// qkv tokens (L*N x 1536) -> q/k packed (n,h,L,64) and v packed transposed (n,h,64,L)
__global__ void k_pack_qkv(const float* __restrict__ qkv,
                           bf16* __restrict__ q, bf16* __restrict__ k, bf16* __restrict__ vt,
                           int L, int N, long total) {
    for (long i = blockIdx.x * 256L + threadIdx.x; i < total; i += gridDim.x * 256L) {
        int d = (int)(i & 63);
        long r = i >> 6;
        int l = (int)(r % L);
        long r2 = r / L;
        int h = (int)(r2 & 7);
        int n = (int)(r2 >> 3);
        const float* base = qkv + ((long)l * N + n) * 1536 + h * 64 + d;
        q[i] = (bf16)base[0];
        k[i] = (bf16)base[512];
        vt[((long)(n * 8 + h) * 64 + d) * L + l] = (bf16)base[1024];
    }
}

// per-(n,h) attention output (batch,L,64) f32 -> token-major (L*N x 512) bf16
__global__ void k_unpack_o(const float* __restrict__ o, bf16* __restrict__ tok,
                           int L, int N, long total) {
    for (long i = blockIdx.x * 256L + threadIdx.x; i < total; i += gridDim.x * 256L) {
        int c = (int)(i & 511);
        long tr = i >> 9;
        int l = (int)(tr / N), n = (int)(tr % N);
        int h = c >> 6, d = c & 63;
        tok[i] = (bf16)o[(((long)(n * 8 + h) * L + l) << 6) + d];
    }
}

// feature map (B,C,32,32) -> horizontal/vertical token matrices (16384 x 512)
__global__ void k_build_hv(const float* __restrict__ fm, float* __restrict__ xf,
                           bf16* __restrict__ xb, int vert, long total) {
    for (long i = blockIdx.x * 256L + threadIdx.x; i < total; i += gridDim.x * 256L) {
        int c = (int)(i & 511);
        long pos = i >> 9;
        int p0 = (int)(pos & 31), p1 = (int)((pos >> 5) & 31), b = (int)(pos >> 10);
        int h = vert ? p0 : p1;
        int w = vert ? p1 : p0;
        float v = fm[(((long)b * 512 + c) * 32 + h) * 32 + w];
        xf[i] = v; xb[i] = (bf16)v;
    }
}

// faithful channel-dim-unfold block tokens: row r = b1*512+b2*256+p*16+q
__global__ void k_build_s(const float* __restrict__ fm, float* __restrict__ xf,
                          bf16* __restrict__ xb, long total) {
    for (long i = blockIdx.x * 256L + threadIdx.x; i < total; i += gridDim.x * 256L) {
        int c2 = (int)(i & 511);
        int r  = (int)((i >> 9) & 1023);
        int b  = (int)(i >> 19);
        int f  = (c2 << 10) + r;
        int cb = f >> 14, hb = (f >> 13) & 1, w = (f >> 8) & 31, ci = (f >> 4) & 15, hi = f & 15;
        int c = cb * 16 + ci, h = hb * 16 + hi;
        float v = fm[(((long)b * 512 + c) * 32 + h) * 32 + w];
        xf[i] = v; xb[i] = (bf16)v;
    }
}

// fused (b, m=h*32+w, k<1024) from H-block / V-block token outputs
__global__ void k_build_fused(const bf16* __restrict__ HB, const bf16* __restrict__ VB,
                              bf16* __restrict__ dst, long total) {
    for (long i = blockIdx.x * 256L + threadIdx.x; i < total; i += gridDim.x * 256L) {
        int k = (int)(i & 1023);
        int m = (int)((i >> 10) & 1023);
        int b = (int)(i >> 20);
        int h = m >> 5, w = m & 31;
        dst[i] = (k < 512) ? HB[((long)b * 1024 + h * 32 + w) * 512 + k]
                           : VB[((long)b * 1024 + w * 32 + h) * 512 + (k - 512)];
    }
}

// gm (b,512,1024) f32 -> position-major (b,1024,512) bf16
__global__ void k_gm_transpose(const float* __restrict__ gm, bf16* __restrict__ dst, long total) {
    for (long i = blockIdx.x * 256L + threadIdx.x; i < total; i += gridDim.x * 256L) {
        int c = (int)(i & 511);
        int m = (int)((i >> 9) & 1023);
        int b = (int)(i >> 19);
        dst[i] = (bf16)gm[((long)b << 19) + ((long)c << 10) + m];
    }
}

// conv2 input: (gm + reassembled att) position-major bf16; faithful att4 raw reshape
__global__ void k_build_conv2_in(const float* __restrict__ GM, const float* __restrict__ ATT,
                                 bf16* __restrict__ CX, long total) {
    for (long i = blockIdx.x * 256L + threadIdx.x; i < total; i += gridDim.x * 256L) {
        int m  = (int)(i & 1023);
        int c2 = (int)((i >> 10) & 511);
        int b  = (int)(i >> 19);
        int h = m >> 5, w = m & 31;
        int b1 = h >> 4, p = h & 15, b2 = w >> 4, q = w & 15;
        int blk = b1 * 2 + b2;
        int g = (c2 << 8) + (p << 4) + q;          // c2*256 + p*16 + q
        int n = g >> 9, cc = g & 511;              // raw reshape (256,C)->(C,16,16)
        float av = ATT[((long)(b * 4 + blk) * 256 + n) * 512 + cc];
        float gv = GM[((long)b << 19) + ((long)c2 << 10) + m];
        CX[((long)b << 19) + (long)m * 512 + c2] = (bf16)(gv + av);
    }
}

// ===========================================================================
// Host orchestration
// ===========================================================================
static inline int nblk(long n) {
    long b = (n + 255) / 256;
    return (int)(b > 262144 ? 262144 : (b < 1 ? 1 : b));
}

extern "C" void kernel_launch(void* const* d_in, const int* in_sizes, int n_in,
                              void* d_out, int out_size, void* d_ws, size_t ws_size,
                              hipStream_t stream)
{
    (void)in_sizes; (void)n_in; (void)out_size; (void)ws_size;
    const float* fm = (const float*)d_in[0];
    const int TOK = 16384;                 // 16*32*32 tokens, C=512

    char* ws = (char*)d_ws;
    size_t off = 0;
    auto alloc = [&](size_t bytes) -> void* {
        void* p = ws + off;
        off += (bytes + 255) & ~(size_t)255;
        return p;
    };

    auto gemm = [&](const bf16* A, long sA, int lda, const bf16* Wt, long sW, int ldw,
                    float* Y, long sY, int ldy, int M, int N, int K, int batch,
                    const float* bias, int bmode, int relu) {
        dim3 g((N + 63) / 64, (M + 127) / 128, batch);
        bool guarded = (M & 127) || (N & 63) || (K & 63);
        if (guarded)
            k_gemm<true><<<g, 256, 0, stream>>>(A, sA, lda, Wt, sW, ldw, Y, sY, ldy,
                                                bias, bmode, relu, M, N, K);
        else
            k_gemm<false><<<g, 256, 0, stream>>>(A, sA, lda, Wt, sW, ldw, Y, sY, ldy,
                                                 bias, bmode, relu, M, N, K);
    };
    auto cvtw = [&](int idx, long n) -> bf16* {
        bf16* p = (bf16*)alloc(n * 2);
        k_f32_to_bf16<<<nblk(n), 256, 0, stream>>>((const float*)d_in[idx], p, n);
        return p;
    };

    // ---- weights -> bf16 (persistent) ----
    bf16 *wq[3], *wo[3], *w1[3], *w2[3];
    const float *bq[3], *bo_[3], *b1_[3], *b2_[3], *g1[3], *be1[3], *g2[3], *be2[3];
    int ffd[3] = {1024, 512, 1024};
    for (int blkI = 0; blkI < 3; ++blkI) {
        int base = 1 + blkI * 12;
        wq[blkI] = cvtw(base + 0, 1536L * 512);
        wo[blkI] = cvtw(base + 2, 512L * 512);
        w1[blkI] = cvtw(base + 4, (long)ffd[blkI] * 512);
        w2[blkI] = cvtw(base + 6, 512L * ffd[blkI]);
        bq[blkI]  = (const float*)d_in[base + 1];
        bo_[blkI] = (const float*)d_in[base + 3];
        b1_[blkI] = (const float*)d_in[base + 5];
        b2_[blkI] = (const float*)d_in[base + 7];
        g1[blkI]  = (const float*)d_in[base + 8];
        be1[blkI] = (const float*)d_in[base + 9];
        g2[blkI]  = (const float*)d_in[base + 10];
        be2[blkI] = (const float*)d_in[base + 11];
    }
    bf16* c1w = cvtw(37, 512L * 1024);
    bf16* c2w = cvtw(39, 512L * 512);
    const float* c1b = (const float*)d_in[38];
    const float* c2b = (const float*)d_in[40];

    // ---- persistent activations ----
    bf16*  HB   = (bf16*)alloc(16777216);       // h-block output tokens (bf16)
    bf16*  VB   = (bf16*)alloc(16777216);
    bf16*  SB   = (bf16*)alloc(16777216);
    float* GM   = (float*)alloc(33554432);      // conv1 output (b,512,1024)
    bf16*  GMMC = (bf16*)alloc(16777216);       // gm position-major
    bf16*  GMCM = (bf16*)alloc(16777216);       // gm channel-major

    // ---- reusable scratch (lifetime-aliased) ----
    float* x0f  = (float*)alloc(33554432);
    bf16*  x0b  = (bf16*)alloc(16777216);
    float* qkvF = (float*)alloc(100663296);     // also ff1F, also cross ATT
    bf16*  qpk  = (bf16*)alloc(16777216);
    bf16*  kpk  = (bf16*)alloc(16777216);
    bf16*  vtpk = (bf16*)alloc(16777216);
    float* scF  = (float*)alloc(268435456);     // scores; also proj; also cross scores
    bf16*  prB  = (bf16*)alloc(134217728);      // probs; also ff1B / FUSED / cross probs
    float* oF   = (float*)alloc(33554432);      // attn out; also ff2F
    bf16*  otB  = (bf16*)alloc(16777216);       // attn tokens; also CX
    float* t1f  = (float*)alloc(33554432);
    bf16*  t1b  = (bf16*)alloc(16777216);

    float* ff1F = qkvF;
    bf16*  ff1B = prB;
    float* projF = scF;
    float* ff2F = oF;

    // ---- one transformer block (faithful _tblock) ----
    auto run_block = [&](int bi, const float* xf, const bf16* xb, int L, int Nb, bf16* out_bf) {
        // QKV projection
        gemm(xb, 0, 512, wq[bi], 0, 512, qkvF, 0, 1536, TOK, 1536, 512, 1, bq[bi], 1, 0);
        // pack per-(n,h) q,k and transposed v
        long tp = (long)Nb * 8 * L * 64;
        k_pack_qkv<<<nblk(tp), 256, 0, stream>>>(qkvF, qpk, kpk, vtpk, L, Nb, tp);
        int batch = Nb * 8;
        // scores = q @ k^T  (M=N=L, K=64)
        gemm(qpk, (long)L * 64, 64, kpk, (long)L * 64, 64,
             scF, (long)L * L, L, L, L, 64, batch, nullptr, 0, 0);
        k_softmax<<<(unsigned)((long)batch * L), 256, 0, stream>>>(scF, prB, L, 0.125f);
        // o = probs @ v  (via vt: M=L, N=64, K=L)
        gemm(prB, (long)L * L, L, vtpk, 64L * L, L,
             oF, (long)L * 64, 64, L, 64, L, batch, nullptr, 0, 0);
        k_unpack_o<<<nblk((long)TOK * 512), 256, 0, stream>>>(oF, otB, L, Nb, (long)TOK * 512);
        // out projection
        gemm(otB, 0, 512, wo[bi], 0, 512, projF, 0, 512, TOK, 512, 512, 1, bo_[bi], 1, 0);
        // x1 = LN(x0 + attn)
        k_add_ln<<<TOK, 256, 0, stream>>>(xf, projF, g1[bi], be1[bi], t1f, t1b, 512);
        // FF
        gemm(t1b, 0, 512, w1[bi], 0, 512, ff1F, 0, ffd[bi], TOK, ffd[bi], 512, 1, b1_[bi], 1, 1);
        k_f32_to_bf16<<<nblk((long)TOK * ffd[bi]), 256, 0, stream>>>(ff1F, ff1B, (long)TOK * ffd[bi]);
        gemm(ff1B, 0, ffd[bi], w2[bi], 0, ffd[bi], ff2F, 0, 512, TOK, 512, ffd[bi], 1, b2_[bi], 1, 0);
        // x2 = LN(x1 + ff)
        k_add_ln<<<TOK, 256, 0, stream>>>(t1f, ff2F, g2[bi], be2[bi], t1f, out_bf, 512);
    };

    long ntok = (long)TOK * 512;

    // ---- horizontal / vertical transformers ----
    k_build_hv<<<nblk(ntok), 256, 0, stream>>>(fm, x0f, x0b, 0, ntok);
    run_block(0, x0f, x0b, 512, 32, HB);
    k_build_hv<<<nblk(ntok), 256, 0, stream>>>(fm, x0f, x0b, 1, ntok);
    run_block(1, x0f, x0b, 512, 32, VB);

    // ---- fuse + conv1 -> gm ----
    bf16* FUSED = prB;                              // 16*1024*1024 bf16 fits in prB
    k_build_fused<<<nblk(16L * 1024 * 1024), 256, 0, stream>>>(HB, VB, FUSED, 16L * 1024 * 1024);
    gemm(c1w, 0, 1024, FUSED, 1024L * 1024, 1024,
         GM, 512L * 1024, 1024, 512, 1024, 1024, 16, c1b, 2, 0);
    k_f32_to_bf16<<<nblk(16L * 512 * 1024), 256, 0, stream>>>(GM, GMCM, 16L * 512 * 1024);
    k_gm_transpose<<<nblk(16L * 1024 * 512), 256, 0, stream>>>(GM, GMMC, 16L * 1024 * 512);

    // ---- self-attention block on scrambled block tokens ----
    k_build_s<<<nblk(ntok), 256, 0, stream>>>(fm, x0f, x0b, ntok);
    run_block(2, x0f, x0b, 16, 1024, SB);

    // ---- cross attention: scores = bq @ gm, softmax, att = aw @ gm^T ----
    gemm(SB, 1024L * 512, 512, GMMC, 1024L * 512, 512,
         scF, 1024L * 1024, 1024, 1024, 1024, 512, 16, nullptr, 0, 0);
    k_softmax<<<16 * 1024, 256, 0, stream>>>(scF, prB, 1024, 1.0f);
    float* ATT = qkvF;                              // (b,1024,512) fits in qkvF
    gemm(prB, 1024L * 1024, 1024, GMCM, 512L * 1024, 1024,
         ATT, 1024L * 512, 512, 1024, 512, 1024, 16, nullptr, 0, 0);

    // ---- reassemble + conv2 -> d_out (b,512,32,32) ----
    bf16* CX = otB;
    k_build_conv2_in<<<nblk(16L * 512 * 1024), 256, 0, stream>>>(GM, ATT, CX, 16L * 512 * 1024);
    gemm(c2w, 0, 512, CX, 1024L * 512, 512,
         (float*)d_out, 512L * 1024, 1024, 512, 1024, 512, 16, c2b, 2, 0);
}
